// MLPBinaryConnect_M1_40252433498722
// MI455X (gfx1250) — compile-verified, matrix operand order checked by reading, not run on hardware
//
#include <hip/hip_runtime.h>

// ---------------------------------------------------------------------------
// Binarized 4-layer MLP: GEMM(sign-weights) -> BatchNorm(batch stats) -> ReLU
// B=16384, IN=784 (pad 800), H=2047 (pad 2048), OUT=10 (pad 64)
// f16 activations / f16 {+1,-1,0} weights, f32 accumulation via
// v_wmma_f32_16x16x32_f16. W tiles double-buffered in LDS via
// global_load_async_to_lds_b128 (ASYNCcnt-tracked DMA, CDNA5).
// Steady-state K loop is branch-free: last iteration peeled.
// ---------------------------------------------------------------------------

typedef __attribute__((ext_vector_type(16))) _Float16 v16h;
typedef __attribute__((ext_vector_type(8)))  float    v8f;

#define BATCH   16384
#define EPS_BN  1e-4f
#define NCHUNK  16          // row chunks for deterministic BN stats
#define ROWS_PER_CHUNK (BATCH / NCHUNK)

// ---------------- pack / binarize (f32 -> padded f16) ----------------------
__global__ __launch_bounds__(256)
void pack_f16(const float* __restrict__ src, _Float16* __restrict__ dst,
              int R, int C, int Rpad, int Cpad, int do_sign)
{
    size_t idx = (size_t)blockIdx.x * 256 + threadIdx.x;
    size_t total = (size_t)Rpad * Cpad;
    if (idx >= total) return;
    int r = (int)(idx / Cpad);
    int c = (int)(idx % Cpad);
    float v = 0.f;
    if (r < R && c < C) {
        v = src[(size_t)r * C + c];
        if (do_sign) v = (v > 0.f) ? 1.f : ((v < 0.f) ? -1.f : 0.f);
    }
    dst[idx] = (_Float16)v;
}

// one 32-wide K step: 2 A fragments x 4 B fragments -> 8 WMMAs
__device__ __forceinline__
void kstep(const _Float16* __restrict__ lwbuf,   // LDS W tile [64 x 32]
           const _Float16* __restrict__ abase,   // A frag base (lane-resolved)
           size_t lda, int lane, v8f acc[2][4])
{
    v16h a0 = *(const v16h*)(abase);
    v16h a1 = *(const v16h*)(abase + 16 * lda);
    const _Float16* lb = lwbuf + (lane & 15) * 32 + ((lane >> 4) * 16);
    #pragma unroll
    for (int s = 0; s < 4; ++s) {
        v16h b = *(const v16h*)(lb + s * 16 * 32);
        acc[0][s] = __builtin_amdgcn_wmma_f32_16x16x32_f16(
            false, a0, false, b, (short)0, acc[0][s], false, false);
        acc[1][s] = __builtin_amdgcn_wmma_f32_16x16x32_f16(
            false, a1, false, b, (short)0, acc[1][s], false, false);
    }
}

// ---------------- GEMM: Y[M x Npad] = A[M x Kpad] * W[Npad x Kpad]^T -------
// block = 256 threads = 8 waves; block tile 256(M) x 64(N); K step 32.
// Each wave owns a 32x64 tile (8 accumulators); W tile [64 x 32] is
// double-buffered in LDS via async global->LDS DMA.
__global__ __launch_bounds__(256)
void gemm_bin_f16(const _Float16* __restrict__ A, int lda,
                  const _Float16* __restrict__ W, int ldw,
                  float* __restrict__ Y, int ldy, int Kpad)
{
    __shared__ __align__(32) _Float16 lw[2][64 * 32];

    const int tid    = threadIdx.x;
    const int lane   = tid & 31;
    const int wave   = tid >> 5;
    const int n_base = blockIdx.x * 64;
    const int m_base = blockIdx.y * 256 + wave * 32;

    // per-thread 16B slice of the [64(N) x 32(K)] W tile
    const int wrow = tid >> 2;                 // 0..63
    const int wseg = (tid & 3) * 8;            // halves 0,8,16,24
    const _Float16* wnext = W + (size_t)(n_base + wrow) * ldw + wseg;
    // raw LDS byte offsets (generic shared address truncated to 32 bits)
    const unsigned ldst0 = (unsigned)(size_t)(&lw[0][wrow * 32 + wseg]);
    const unsigned ldst1 = (unsigned)(size_t)(&lw[1][wrow * 32 + wseg]);

    // A fragment base: lane%16 -> row, lane/16 -> K half (32B per lane)
    const _Float16* abase =
        A + (size_t)(m_base + (lane & 15)) * lda + ((lane >> 4) * 16);

    v8f acc[2][4];
    #pragma unroll
    for (int h = 0; h < 2; ++h)
        #pragma unroll
        for (int s = 0; s < 4; ++s) {
            v8f z = {0.f, 0.f, 0.f, 0.f, 0.f, 0.f, 0.f, 0.f};
            acc[h][s] = z;
        }

    const int ntiles = Kpad >> 5;

    // preload tile 0 into LDS buffer 0 (async DMA, ASYNCcnt-tracked)
    asm volatile("global_load_async_to_lds_b128 %0, %1, off"
                 :: "v"(ldst0), "v"(wnext) : "memory");
    wnext += 32;

    // steady state: branch-free; always one DMA in flight
    for (int it = 0; it < ntiles - 1; ++it) {
        const unsigned dst = ((it + 1) & 1) ? ldst1 : ldst0;
        asm volatile("global_load_async_to_lds_b128 %0, %1, off"
                     :: "v"(dst), "v"(wnext) : "memory");
        wnext += 32;
        // older async load (current tile) complete when count <= 1
        asm volatile("s_wait_asynccnt 0x1" ::: "memory");
        __syncthreads();

        __builtin_prefetch(abase + 32, 0, 3);
        __builtin_prefetch(abase + 16 * (size_t)lda + 32, 0, 3);
        kstep(lw[it & 1], abase, (size_t)lda, lane, acc);
        abase += 32;
        __syncthreads();   // protect LDS buffer being DMA'd next iteration
    }

    // epilogue: last tile
    asm volatile("s_wait_asynccnt 0x0" ::: "memory");
    __syncthreads();
    kstep(lw[(ntiles - 1) & 1], abase, (size_t)lda, lane, acc);

    // C layout: VGPR j -> M = j + 8*(lane>=16), N = lane%16
    const int crow = m_base + ((lane >> 4) << 3);
    const int ccol = lane & 15;
    #pragma unroll
    for (int h = 0; h < 2; ++h)
        #pragma unroll
        for (int s = 0; s < 4; ++s) {
            float* yp = Y + (size_t)(crow + 16 * h) * ldy + n_base + s * 16 + ccol;
            #pragma unroll
            for (int j = 0; j < 8; ++j)
                yp[(size_t)j * ldy] = acc[h][s][j];
        }
}

// ---------------- BN stats: deterministic two-level column reduction -------
__global__ __launch_bounds__(256)
void stats_partial(const float* __restrict__ Y, int ldy, int Npad,
                   float* __restrict__ part1, float* __restrict__ part2)
{
    const int col   = blockIdx.x * 256 + threadIdx.x;
    const int chunk = blockIdx.y;
    if (col >= Npad) return;
    float s1 = 0.f, s2 = 0.f;
    const int r0 = chunk * ROWS_PER_CHUNK;
    for (int r = r0; r < r0 + ROWS_PER_CHUNK; ++r) {
        float v = Y[(size_t)r * ldy + col];
        s1 += v;
        s2 += v * v;
    }
    part1[(size_t)chunk * Npad + col] = s1;
    part2[(size_t)chunk * Npad + col] = s2;
}

__global__ __launch_bounds__(256)
void stats_final(const float* __restrict__ part1, const float* __restrict__ part2,
                 const float* __restrict__ g, const float* __restrict__ b,
                 int Nreal, int Npad,
                 float* __restrict__ coefA, float* __restrict__ coefB)
{
    const int n = blockIdx.x * 256 + threadIdx.x;
    if (n >= Npad) return;
    float Ac = 0.f, Bc = 0.f;
    if (n < Nreal) {
        float s1 = 0.f, s2 = 0.f;
        for (int c = 0; c < NCHUNK; ++c) {
            s1 += part1[(size_t)c * Npad + n];
            s2 += part2[(size_t)c * Npad + n];
        }
        const float invB = 1.f / (float)BATCH;
        float mu  = s1 * invB;
        float var = s2 * invB - mu * mu;
        float rs  = rsqrtf(var + EPS_BN);
        Ac = rs * g[n];
        Bc = b[n] - mu * Ac;
    }
    coefA[n] = Ac;
    coefB[n] = Bc;
}

// ---------------- BN apply + ReLU -> f16 activations (in place ping) -------
__global__ __launch_bounds__(256)
void apply_act(const float* __restrict__ Y, int Npad,
               const float* __restrict__ coefA, const float* __restrict__ coefB,
               _Float16* __restrict__ act)
{
    size_t idx = (size_t)blockIdx.x * 256 + threadIdx.x;
    if (idx >= (size_t)BATCH * Npad) return;
    int n = (int)(idx % Npad);
    float v = Y[idx] * coefA[n] + coefB[n];
    v = (v > 0.f) ? v : 0.f;
    act[idx] = (_Float16)v;
}

// ---------------- final BN (no ReLU) -> f32 d_out [16384 x 10] -------------
__global__ __launch_bounds__(256)
void apply_out(const float* __restrict__ Y, int ldy,
               const float* __restrict__ coefA, const float* __restrict__ coefB,
               float* __restrict__ out)
{
    size_t idx = (size_t)blockIdx.x * 256 + threadIdx.x;
    if (idx >= (size_t)BATCH * 10) return;
    int row = (int)(idx / 10);
    int n   = (int)(idx % 10);
    out[idx] = Y[(size_t)row * ldy + n] * coefA[n] + coefB[n];
}

// ---------------------------------------------------------------------------
extern "C" void kernel_launch(void* const* d_in, const int* in_sizes, int n_in,
                              void* d_out, int out_size, void* d_ws, size_t ws_size,
                              hipStream_t stream)
{
    const float* x  = (const float*)d_in[0];
    const float* w1 = (const float*)d_in[1];
    const float* w2 = (const float*)d_in[2];
    const float* w3 = (const float*)d_in[3];
    const float* wm = (const float*)d_in[4];
    const float* g1 = (const float*)d_in[5];  const float* b1 = (const float*)d_in[6];
    const float* g2 = (const float*)d_in[7];  const float* b2 = (const float*)d_in[8];
    const float* g3 = (const float*)d_in[9];  const float* b3 = (const float*)d_in[10];
    const float* g4 = (const float*)d_in[11]; const float* b4 = (const float*)d_in[12];
    float* out = (float*)d_out;

    // ---- workspace carve-out (aligned 256B) ----
    char* ws = (char*)d_ws;
    size_t off = 0;
    auto take = [&](size_t bytes) -> char* {
        char* p = ws + off;
        off += (bytes + 255) & ~(size_t)255;
        return p;
    };
    float*    Yf   = (float*)   take((size_t)BATCH * 2048 * 4);   // 128 MB
    _Float16* act  = (_Float16*)take((size_t)BATCH * 2048 * 2);   //  64 MB
    _Float16* W1   = (_Float16*)take((size_t)2048 * 800 * 2);
    _Float16* W2   = (_Float16*)take((size_t)2048 * 2048 * 2);
    _Float16* W3   = (_Float16*)take((size_t)2048 * 2048 * 2);
    _Float16* WM   = (_Float16*)take((size_t)64 * 2048 * 2);
    float*    p1   = (float*)   take((size_t)NCHUNK * 2048 * 4);
    float*    p2   = (float*)   take((size_t)NCHUNK * 2048 * 4);
    float*    cA   = (float*)   take((size_t)2048 * 4);
    float*    cB   = (float*)   take((size_t)2048 * 4);

    auto blocks = [](size_t total) { return (unsigned)((total + 255) / 256); };

    // ---- pack inputs ----
    pack_f16<<<blocks((size_t)BATCH * 800), 256, 0, stream>>>(x,  act, BATCH, 784, BATCH, 800, 0);
    pack_f16<<<blocks((size_t)2048 * 800), 256, 0, stream>>>(w1, W1, 2047, 784, 2048, 800, 1);
    pack_f16<<<blocks((size_t)2048 * 2048), 256, 0, stream>>>(w2, W2, 2047, 2047, 2048, 2048, 1);
    pack_f16<<<blocks((size_t)2048 * 2048), 256, 0, stream>>>(w3, W3, 2047, 2047, 2048, 2048, 1);
    pack_f16<<<blocks((size_t)64 * 2048), 256, 0, stream>>>(wm, WM, 10, 2047, 64, 2048, 1);

    // ---- layers 1..3: GEMM -> stats -> BN+ReLU (f16, in-place ping) ----
    struct Layer { const _Float16* W; int lda, Kpad; const float *g, *b; };
    Layer L[3] = {
        { W1,  800,  800, g1, b1 },
        { W2, 2048, 2048, g2, b2 },
        { W3, 2048, 2048, g3, b3 },
    };
    for (int li = 0; li < 3; ++li) {
        gemm_bin_f16<<<dim3(2048 / 64, BATCH / 256), 256, 0, stream>>>(
            act, L[li].lda, L[li].W, L[li].Kpad, Yf, 2048, L[li].Kpad);
        stats_partial<<<dim3(2048 / 256, NCHUNK), 256, 0, stream>>>(Yf, 2048, 2048, p1, p2);
        stats_final<<<2048 / 256, 256, 0, stream>>>(p1, p2, L[li].g, L[li].b, 2047, 2048, cA, cB);
        apply_act<<<blocks((size_t)BATCH * 2048), 256, 0, stream>>>(Yf, 2048, cA, cB, act);
    }

    // ---- layer 4: GEMM (N padded 10->64) -> stats -> BN -> d_out ----
    gemm_bin_f16<<<dim3(1, BATCH / 256), 256, 0, stream>>>(
        act, 2048, WM, 2048, Yf, 64, 2048);
    stats_partial<<<dim3(1, NCHUNK), 256, 0, stream>>>(Yf, 64, 64, p1, p2);
    stats_final<<<1, 256, 0, stream>>>(p1, p2, g4, b4, 10, 64, cA, cB);
    apply_out<<<blocks((size_t)BATCH * 10), 256, 0, stream>>>(Yf, 64, cA, cB, out);
}
